// ActionLayer_17205638988619
// MI455X (gfx1250) — matrix operand order repaired and checked by reading, not compiled, not run on hardware
//
#include <hip/hip_runtime.h>
#include <hip/hip_bf16.h>
#include <math.h>

// ---------------------------------------------------------------------------
// Problem constants (from reference setup_inputs)
// ---------------------------------------------------------------------------
constexpr int B_    = 2;
constexpr int T_    = 1024;
constexpr int V_    = 32000;
constexpr int H_    = 1024;
constexpr int BT    = B_ * T_;     // 2048 positions
constexpr int K2    = 2 * H_;      // 2048 fused input dim
constexpr int INNER = 2 * H_;      // 2048 MLP inner dim
constexpr int TOPK  = 16;
constexpr int NACT  = 17;          // topk + label
constexpr int AROWS = 32;          // padded action rows (2 WMMA M-tiles)

typedef __bf16 v16bf __attribute__((ext_vector_type(16)));
typedef float  v8f   __attribute__((ext_vector_type(8)));

union Frag16 {
    uint4 q[2];
    v16bf v;
    unsigned short s[16];
};

struct RowMeta { float logZ; int denom; int lblMask; int valid; };

__device__ __forceinline__ unsigned short f2bf(float f) {
    unsigned int u = __float_as_uint(f);
    unsigned int r = u + 0x7FFFu + ((u >> 16) & 1u);   // round-to-nearest-even
    return (unsigned short)(r >> 16);
}

__device__ __forceinline__ float geluf(float x) {
    return 0.5f * x * (1.0f + erff(x * 0.70710678118654752f));
}

// ---------------------------------------------------------------------------
// Kernel 0: zero the loss / valid-count accumulators (graph-replay safe)
// ---------------------------------------------------------------------------
__global__ void init_kernel(float* accums) {
    if (threadIdx.x < 2) accums[threadIdx.x] = 0.0f;
}

// ---------------------------------------------------------------------------
// Kernel 1: transpose + convert W1 (2048x2048) and W2 (2048x1024) to bf16
// in [N][K] layout so B-fragments are contiguous 32B per lane.
// ---------------------------------------------------------------------------
__global__ __launch_bounds__(256) void convert_kernel(
        const float* __restrict__ W1, const float* __restrict__ W2,
        unsigned short* __restrict__ W1t, unsigned short* __restrict__ W2t) {
    int e = blockIdx.x * 256 + threadIdx.x;
    const int n1 = K2 * INNER;            // 4194304
    const int n2 = INNER * H_;            // 2097152
    if (e < n1) {
        int f = e / INNER, i = e - f * INNER;     // W1[f][i]
        W1t[(size_t)i * K2 + f] = f2bf(W1[e]);
    } else if (e < n1 + n2) {
        int e2 = e - n1;
        int i = e2 / H_, h = e2 - i * H_;         // W2[i][h]
        W2t[(size_t)h * INNER + i] = f2bf(W2[e2]);
    }
}

// ---------------------------------------------------------------------------
// Kernel 2: per-row top-16 + label action set, log-sum-exp of logits row,
// valid count.  One 256-thread block per row.
// ---------------------------------------------------------------------------
__global__ __launch_bounds__(256) void topk_kernel(
        const float* __restrict__ logits, const int* __restrict__ labels,
        const unsigned char* __restrict__ fvalid,
        int* __restrict__ actIdx, RowMeta* __restrict__ meta,
        float* __restrict__ accums) {
    __shared__ float lv[4096];
    __shared__ int   li[4096];
    __shared__ float rv[8];
    __shared__ int   ri[8];
    __shared__ int   sel[TOPK];
    __shared__ float selv[TOPK];
    __shared__ float sred[8];

    const int row  = blockIdx.x;
    const int tid  = threadIdx.x;
    const int lane = tid & 31;
    const int wv   = tid >> 5;
    const float* lp = logits + (size_t)row * V_;

    // per-thread register top-16 over a strided 125-element slice
    float tv[TOPK]; int ti[TOPK];
#pragma unroll
    for (int j = 0; j < TOPK; ++j) { tv[j] = -3.4e38f; ti[j] = 0; }
    for (int j = 0; j < V_ / 256; ++j) {
        int idx = tid + j * 256;
        float v = lp[idx];
        if (v > tv[TOPK - 1]) {
            tv[TOPK - 1] = v; ti[TOPK - 1] = idx;
#pragma unroll
            for (int k = TOPK - 1; k > 0; --k) {
                if (tv[k] > tv[k - 1]) {
                    float tf = tv[k]; tv[k] = tv[k - 1]; tv[k - 1] = tf;
                    int   td = ti[k]; ti[k] = ti[k - 1]; ti[k - 1] = td;
                }
            }
        }
    }
#pragma unroll
    for (int j = 0; j < TOPK; ++j) { lv[tid * TOPK + j] = tv[j]; li[tid * TOPK + j] = ti[j]; }
    __syncthreads();

    // 16 cooperative argmax passes over the 4096 candidates
    for (int p = 0; p < TOPK; ++p) {
        float bv = -3.4e38f; int bp = 0;
        for (int j = tid; j < 4096; j += 256) {
            float v = lv[j];
            if (v > bv) { bv = v; bp = j; }
        }
        for (int off = 16; off > 0; off >>= 1) {
            float ov = __shfl_down(bv, off, 32);
            int   op = __shfl_down(bp, off, 32);
            if (ov > bv) { bv = ov; bp = op; }
        }
        if (lane == 0) { rv[wv] = bv; ri[wv] = bp; }
        __syncthreads();
        if (tid == 0) {
            float best = rv[0]; int bpos = ri[0];
            for (int w = 1; w < 8; ++w) if (rv[w] > best) { best = rv[w]; bpos = ri[w]; }
            sel[p] = li[bpos]; selv[p] = best; lv[bpos] = -3.4e38f;
        }
        __syncthreads();
    }

    // log-sum-exp with rowmax = top-1
    float rowmax = selv[0];
    float se = 0.0f;
    for (int j = 0; j < V_ / 256; ++j) se += expf(lp[tid + j * 256] - rowmax);
    for (int off = 16; off > 0; off >>= 1) se += __shfl_down(se, off, 32);
    if (lane == 0) sred[wv] = se;
    __syncthreads();

    if (tid == 0) {
        float s = 0.0f;
        for (int w = 0; w < 8; ++w) s += sred[w];
        float logZ = rowmax + logf(s);
        int label = labels[row];
        int found = 0;
        for (int j = 0; j < TOPK; ++j) {
            actIdx[row * 32 + j] = sel[j];
            if (sel[j] == label) found = 1;
        }
        actIdx[row * 32 + TOPK] = (label >= 0) ? label : 0;
        RowMeta m;
        m.logZ    = logZ;
        m.lblMask = found ? 0 : 1;
        m.denom   = TOPK + m.lblMask;
        m.valid   = (label != -100) && (fvalid[row] != 0);
        meta[row] = m;
        if (m.valid) atomicAdd(&accums[1], 1.0f);
    }
}

// ---------------------------------------------------------------------------
// Kernel 3: fused per-position MLP (bf16 WMMA) + LayerNorm + cosine +
// softmax rewards + loss accumulation.  One workgroup (8 wave32) per
// position; 256 KB of CDNA5 LDS: A(32x2048 bf16) | h(32x2048 bf16),
// delta(32x1024 f32) aliases A.  Each wave owns PAIRS of N-tiles so each
// K-chunk issues 4 WMMAs against 4 LDS + 4 global b128 loads.
// ---------------------------------------------------------------------------
__global__ __launch_bounds__(256) void fused_kernel(
        const float* __restrict__ logits, const float* __restrict__ hidden,
        const float* __restrict__ future, const float* __restrict__ embed,
        const float* __restrict__ b1, const float* __restrict__ b2,
        const float* __restrict__ lng, const float* __restrict__ lnb,
        const unsigned short* __restrict__ W1t,
        const unsigned short* __restrict__ W2t,
        const int* __restrict__ actIdx, const RowMeta* __restrict__ meta,
        float* __restrict__ accums) {
    extern __shared__ char smem[];
    unsigned short* bfA    = (unsigned short*)smem;              // 32x2048 bf16 (128KB)
    unsigned short* bfH    = (unsigned short*)(smem + 131072);   // 32x2048 bf16 (128KB)
    float*          deltaF = (float*)smem;                       // 32x1024 f32, aliases bfA
    float*          scores = (float*)(smem + 262144);            // 32 floats

    const int p    = blockIdx.x;
    const int tid  = threadIdx.x;
    const int lane = tid & 31;
    const int wv   = tid >> 5;
    const RowMeta m = meta[p];
    const int* aidx = actIdx + p * 32;

    // ---- build A: rows 0..15 topk, row 16 label (if unique), rest zero ----
    for (int e = tid; e < AROWS * K2; e += 256) {
        int r = e >> 11, c = e & (K2 - 1);
        float v = 0.0f;
        if (r < NACT && (r < TOPK || m.lblMask)) {
            if (c < H_) v = hidden[(size_t)p * H_ + c];
            else        v = embed[(size_t)aidx[r] * H_ + (c - H_)];
        }
        bfA[e] = f2bf(v);
    }
    __syncthreads();

    const int col16 = lane & 15;
    const int ksel  = lane >> 4;       // 0: K 0..15-half, 1: K 16..31-half

    // ---- GEMM1: h = gelu(A @ W1 + b1), h -> LDS bf16 ----
    for (int ntp = wv; ntp < INNER / 32; ntp += 8) {
        const int nt0 = ntp * 2, nt1 = nt0 + 1;
        v8f acc00 = {}; v8f acc10 = {};   // N-tile 0, M-tiles 0/1
        v8f acc01 = {}; v8f acc11 = {};   // N-tile 1, M-tiles 0/1
        const unsigned short* bp0 = W1t + (size_t)(nt0 * 16 + col16) * K2 + ksel * 16;
        const unsigned short* bp1 = W1t + (size_t)(nt1 * 16 + col16) * K2 + ksel * 16;
        for (int kc = 0; kc < K2; kc += 32) {
            Frag16 bf0, bf1;
            bf0.q[0] = *(const uint4*)(bp0 + kc);
            bf0.q[1] = *(const uint4*)(bp0 + kc + 8);
            bf1.q[0] = *(const uint4*)(bp1 + kc);
            bf1.q[1] = *(const uint4*)(bp1 + kc + 8);
            __builtin_prefetch(bp0 + kc + 64, 0, 3);
            __builtin_prefetch(bp1 + kc + 64, 0, 3);
            Frag16 a0, a1;
            const unsigned short* ap0 = bfA + (size_t)col16 * K2 + kc + ksel * 16;
            a0.q[0] = *(const uint4*)ap0;
            a0.q[1] = *(const uint4*)(ap0 + 8);
            const unsigned short* ap1 = ap0 + 16 * K2;
            a1.q[0] = *(const uint4*)ap1;
            a1.q[1] = *(const uint4*)(ap1 + 8);
            acc00 = __builtin_amdgcn_wmma_f32_16x16x32_bf16(false, a0.v, false, bf0.v,
                                                            (short)0, acc00, false, false);
            acc10 = __builtin_amdgcn_wmma_f32_16x16x32_bf16(false, a1.v, false, bf0.v,
                                                            (short)0, acc10, false, false);
            acc01 = __builtin_amdgcn_wmma_f32_16x16x32_bf16(false, a0.v, false, bf1.v,
                                                            (short)0, acc01, false, false);
            acc11 = __builtin_amdgcn_wmma_f32_16x16x32_bf16(false, a1.v, false, bf1.v,
                                                            (short)0, acc11, false, false);
        }
        const int rbase = ksel * 8;
        const int g0 = nt0 * 16 + col16, g1 = nt1 * 16 + col16;
        const float bias0 = b1[g0], bias1 = b1[g1];
#pragma unroll
        for (int v = 0; v < 8; ++v) {
            bfH[(size_t)(rbase + v) * K2 + g0]      = f2bf(geluf(acc00[v] + bias0));
            bfH[(size_t)(rbase + v + 16) * K2 + g0] = f2bf(geluf(acc10[v] + bias0));
            bfH[(size_t)(rbase + v) * K2 + g1]      = f2bf(geluf(acc01[v] + bias1));
            bfH[(size_t)(rbase + v + 16) * K2 + g1] = f2bf(geluf(acc11[v] + bias1));
        }
    }
    __syncthreads();

    // ---- GEMM2: delta = h @ W2 + b2, delta -> LDS f32 (aliases bfA) ----
    for (int ntp = wv; ntp < H_ / 32; ntp += 8) {
        const int nt0 = ntp * 2, nt1 = nt0 + 1;
        v8f acc00 = {}; v8f acc10 = {};
        v8f acc01 = {}; v8f acc11 = {};
        const unsigned short* bp0 = W2t + (size_t)(nt0 * 16 + col16) * INNER + ksel * 16;
        const unsigned short* bp1 = W2t + (size_t)(nt1 * 16 + col16) * INNER + ksel * 16;
        for (int kc = 0; kc < INNER; kc += 32) {
            Frag16 bf0, bf1;
            bf0.q[0] = *(const uint4*)(bp0 + kc);
            bf0.q[1] = *(const uint4*)(bp0 + kc + 8);
            bf1.q[0] = *(const uint4*)(bp1 + kc);
            bf1.q[1] = *(const uint4*)(bp1 + kc + 8);
            __builtin_prefetch(bp0 + kc + 64, 0, 3);
            __builtin_prefetch(bp1 + kc + 64, 0, 3);
            Frag16 a0, a1;
            const unsigned short* ap0 = bfH + (size_t)col16 * INNER + kc + ksel * 16;
            a0.q[0] = *(const uint4*)ap0;
            a0.q[1] = *(const uint4*)(ap0 + 8);
            const unsigned short* ap1 = ap0 + 16 * INNER;
            a1.q[0] = *(const uint4*)ap1;
            a1.q[1] = *(const uint4*)(ap1 + 8);
            acc00 = __builtin_amdgcn_wmma_f32_16x16x32_bf16(false, a0.v, false, bf0.v,
                                                            (short)0, acc00, false, false);
            acc10 = __builtin_amdgcn_wmma_f32_16x16x32_bf16(false, a1.v, false, bf0.v,
                                                            (short)0, acc10, false, false);
            acc01 = __builtin_amdgcn_wmma_f32_16x16x32_bf16(false, a0.v, false, bf1.v,
                                                            (short)0, acc01, false, false);
            acc11 = __builtin_amdgcn_wmma_f32_16x16x32_bf16(false, a1.v, false, bf1.v,
                                                            (short)0, acc11, false, false);
        }
        const int rbase = ksel * 8;
        const int g0 = nt0 * 16 + col16, g1 = nt1 * 16 + col16;
        const float bias0 = b2[g0], bias1 = b2[g1];
#pragma unroll
        for (int v = 0; v < 8; ++v) {
            deltaF[(size_t)(rbase + v) * H_ + g0]      = acc00[v] + bias0;
            deltaF[(size_t)(rbase + v + 16) * H_ + g0] = acc10[v] + bias0;
            deltaF[(size_t)(rbase + v) * H_ + g1]      = acc01[v] + bias1;
            deltaF[(size_t)(rbase + v + 16) * H_ + g1] = acc11[v] + bias1;
        }
    }
    __syncthreads();

    // ---- LayerNorm + cosine score, one wave per action row ----
    const float* fut = future + (size_t)p * H_;
    for (int r = wv; r < NACT; r += 8) {
        const float* dr = deltaF + (size_t)r * H_;
        float s = 0.0f, s2 = 0.0f;
        for (int c = lane; c < H_; c += 32) { float x = dr[c]; s += x; s2 += x * x; }
        for (int off = 16; off > 0; off >>= 1) {
            s  += __shfl_xor(s,  off, 32);
            s2 += __shfl_xor(s2, off, 32);
        }
        float mu   = s / (float)H_;
        float var  = s2 / (float)H_ - mu * mu;
        float rstd = rsqrtf(var + 1e-5f);
        float dt = 0.0f, na = 0.0f, nf = 0.0f;
        for (int c = lane; c < H_; c += 32) {
            float a = (dr[c] - mu) * rstd * lng[c] + lnb[c];
            float f = fut[c];
            dt += a * f; na += a * a; nf += f * f;
        }
        for (int off = 16; off > 0; off >>= 1) {
            dt += __shfl_xor(dt, off, 32);
            na += __shfl_xor(na, off, 32);
            nf += __shfl_xor(nf, off, 32);
        }
        if (lane == 0)
            scores[r] = dt / (fmaxf(sqrtf(na), 1e-12f) * fmaxf(sqrtf(nf), 1e-12f));
    }
    __syncthreads();

    // ---- softmax rewards + loss (wave 0, one lane per action) ----
    if (wv == 0) {
        bool act = (lane < NACT) && (lane < TOPK || m.lblMask);
        float sc = act ? scores[lane] : -1e30f;
        float mx = sc;
        for (int off = 16; off > 0; off >>= 1) mx = fmaxf(mx, __shfl_xor(mx, off, 32));
        float e = act ? expf(sc - mx) : 0.0f;      // TAU == 1
        float es = e;
        for (int off = 16; off > 0; off >>= 1) es += __shfl_xor(es, off, 32);
        float rw = (es > 0.0f) ? (e / es) : 0.0f;
        float lpa = 0.0f;
        if (act) lpa = logits[(size_t)p * V_ + aidx[lane]] - m.logZ;
        float part = (act && m.valid) ? (rw * lpa) : 0.0f;
        for (int off = 16; off > 0; off >>= 1) part += __shfl_xor(part, off, 32);
        if (lane == 0 && m.valid)
            atomicAdd(&accums[0], -part / (float)m.denom);
    }
}

// ---------------------------------------------------------------------------
// Kernel 4: final normalization
// ---------------------------------------------------------------------------
__global__ void final_kernel(const float* accums, float* out) {
    out[0] = accums[0] / fmaxf(accums[1], 1.0f);
}

// ---------------------------------------------------------------------------
// Launch
// ---------------------------------------------------------------------------
extern "C" void kernel_launch(void* const* d_in, const int* in_sizes, int n_in,
                              void* d_out, int out_size, void* d_ws, size_t ws_size,
                              hipStream_t stream) {
    (void)in_sizes; (void)n_in; (void)out_size; (void)ws_size;
    const float*         logits = (const float*)d_in[0];
    const float*         hidden = (const float*)d_in[1];
    const int*           labels = (const int*)d_in[2];
    const float*         future = (const float*)d_in[3];
    const unsigned char* fvalid = (const unsigned char*)d_in[4];
    const float*         embed  = (const float*)d_in[5];
    const float*         W1     = (const float*)d_in[6];
    const float*         b1     = (const float*)d_in[7];
    const float*         W2     = (const float*)d_in[8];
    const float*         b2     = (const float*)d_in[9];
    const float*         lng    = (const float*)d_in[10];
    const float*         lnb    = (const float*)d_in[11];

    // workspace layout
    char* ws = (char*)d_ws;
    float*          accums = (float*)ws;                               // [0]=loss sum, [1]=valid cnt
    int*            actIdx = (int*)(ws + 256);                         // BT*32 ints
    RowMeta*        meta   = (RowMeta*)(ws + 256 + (size_t)BT * 32 * 4);
    size_t off_w1t = 256 + (size_t)BT * 32 * 4 + (size_t)BT * sizeof(RowMeta); // 295168
    unsigned short* W1t = (unsigned short*)(ws + off_w1t);             // 2048*2048 bf16
    unsigned short* W2t = (unsigned short*)(ws + off_w1t + (size_t)INNER * K2 * 2);

    init_kernel<<<1, 32, 0, stream>>>(accums);

    int convN = K2 * INNER + INNER * H_;
    convert_kernel<<<(convN + 255) / 256, 256, 0, stream>>>(W1, W2, W1t, W2t);

    topk_kernel<<<BT, 256, 0, stream>>>(logits, labels, fvalid, actIdx, meta, accums);

    size_t lds_bytes = 262144 + 256;   // A/delta + h + scores (CDNA5: 320KB LDS/WGP)
    fused_kernel<<<BT, 256, lds_bytes, stream>>>(logits, hidden, future, embed,
                                                 b1, b2, lng, lnb, W1t, W2t,
                                                 actIdx, meta, accums);

    final_kernel<<<1, 1, 0, stream>>>(accums, (float*)d_out);
}
